// TensorFieldNetwork_75531294867758
// MI455X (gfx1250) — matrix operand order repaired and checked by reading, not compiled, not run on hardware
//
#include <hip/hip_runtime.h>
#include <hip/hip_bf16.h>
#include <math.h>

// ---------------------------------------------------------------------------
// TFN classifier, CDNA5 (gfx1250) WMMA implementation.
// Heavy per-pair filter GEMMs are factorized:  msg = (mask*h[,*r]) @ G,
// where G = f_in (x) W2 is precomputed per layer.  All GEMMs run as
// v_wmma_f32_16x16x32_f16 (f16 in, f32 accumulate), tiles staged via LDS.
// Each wave owns one 16-row M tile and ALL N tiles (multi-accumulator),
// so the masked A tile is built once per K step and reused across WMMAs.
// ---------------------------------------------------------------------------

typedef _Float16 half_t;
typedef __attribute__((ext_vector_type(16))) _Float16 v16h;
typedef __attribute__((ext_vector_type(8)))  float    v8f;

#define BATCH  2
#define NATOM  96
#define NPAIR  (NATOM*NATOM)      // 9216
#define NRBF   16
#define KH     32                 // filter hidden width
#define H0C    64
#define H1C    16
#define KJK    (NATOM*KH)         // 3072 contraction rows (j,k)
#define KB00   (KJK + NATOM)      // 3168 (+bias rows)
#define KB10   (NATOM*KH*3 + NATOM*3)  // 9504 (j,k,x)+(j,x) bias
#define RBF_WIDTH_INV (1.0f/0.09765625f)   // 1/((5/16)^2)

__device__ __forceinline__ float fast_rcp(float x) {
    return __builtin_amdgcn_rcpf(x);          // v_rcp_f32 (no NR refine)
}

// ---------------------------------------------------------------------------
// Geometry: dist, r_hat, RBF (f16 for WMMA A), f0/f1 init.
// ---------------------------------------------------------------------------
__global__ void geom_kernel(const float* __restrict__ pos,
                            float* __restrict__ rhat,
                            half_t* __restrict__ rbf,
                            float* __restrict__ f0,
                            float* __restrict__ f1)
{
    int p = blockIdx.x * blockDim.x + threadIdx.x;
    if (p >= BATCH * NPAIR) return;
    int b = p / NPAIR, q = p % NPAIR;
    int i = q / NATOM, j = q % NATOM;
    const float* pi = pos + (size_t)(b*NATOM + i)*3;
    const float* pj = pos + (size_t)(b*NATOM + j)*3;
    float dx = pi[0]-pj[0], dy = pi[1]-pj[1], dz = pi[2]-pj[2];
    float d  = sqrtf(dx*dx + dy*dy + dz*dz);
    float inv = fast_rcp(fmaxf(d, 1e-8f));
    rhat[(size_t)p*3+0] = dx*inv;
    rhat[(size_t)p*3+1] = dy*inv;
    rhat[(size_t)p*3+2] = dz*inv;
#pragma unroll
    for (int r = 0; r < NRBF; ++r) {
        float c = 5.0f * (float)r / 15.0f;
        float t = d - c;
        rbf[(size_t)p*NRBF + r] = (half_t)__expf(-t*t*RBF_WIDTH_INV);
    }
    if (j == 0) {
        f0[(size_t)b*NATOM*H0C + i] =
            sqrtf(pi[0]*pi[0] + pi[1]*pi[1] + pi[2]*pi[2]);
        float* f1r = f1 + (size_t)b*NATOM*H1C*3 + (size_t)i*3;  // in1=1
        f1r[0] = pi[0]; f1r[1] = pi[1]; f1r[2] = pi[2];
    }
}

// ---------------------------------------------------------------------------
// Filter hidden layer: H = silu(RBF @ W1 + b1).  M=B*NPAIR, K=16(pad 32), N=32.
// One wave per 16-row tile; both 16-col tiles kept in registers (2 WMMAs,
// A fragment reused).
// ---------------------------------------------------------------------------
__global__ __launch_bounds__(32) void filter_h_kernel(
    const half_t* __restrict__ rbf, const float* __restrict__ w1,
    const float* __restrict__ b1, half_t* __restrict__ h)
{
    __shared__ half_t As[16][32];
    __shared__ half_t Bs[32][32];
    int lane = threadIdx.x;
    int mt = blockIdx.x;    // 0..B*NPAIR/16-1
#pragma unroll
    for (int e = 0; e < 16; ++e) {
        int q = lane*16 + e;
        int r = q >> 5, k = q & 31;
        As[r][k] = (k < NRBF) ? rbf[(size_t)(mt*16 + r)*NRBF + k] : (half_t)0.0f;
    }
#pragma unroll
    for (int e = 0; e < 32; ++e) {
        int q = lane*32 + e;
        int k = q >> 5, n = q & 31;
        Bs[k][n] = (k < NRBF) ? (half_t)w1[k*KH + n] : (half_t)0.0f;
    }
    __syncthreads();
    int m = lane & 15, hi = lane >> 4;
    v16h a, bf0, bf1;
#pragma unroll
    for (int e = 0; e < 16; ++e) {              // A 16x32, contiguous k = 16*hi + e
        int k = e + 16*hi;
        a[e] = As[m][k];
    }
#pragma unroll
    for (int e = 0; e < 16; ++e) {              // B 32x16, contiguous k = 16*hi + e (same as A)
        bf0[e] = Bs[e + 16*hi][m];
        bf1[e] = Bs[e + 16*hi][16 + m];
    }
    v8f c0 = {}, c1 = {};
    c0 = __builtin_amdgcn_wmma_f32_16x16x32_f16(false, a, false, bf0,
                                                (short)0, c0, false, false);
    c1 = __builtin_amdgcn_wmma_f32_16x16x32_f16(false, a, false, bf1,
                                                (short)0, c1, false, false);
    float bias0 = b1[m], bias1 = b1[16 + m];
#pragma unroll
    for (int v = 0; v < 8; ++v) {
        int row = mt*16 + v + 8*hi;
        float x0 = c0[v] + bias0;
        float x1 = c1[v] + bias1;
        float s0 = x0 * fast_rcp(1.0f + __expf(-x0));     // silu (v_rcp_f32)
        float s1 = x1 * fast_rcp(1.0f + __expf(-x1));
        h[(size_t)row*KH + m]      = (half_t)s0;
        h[(size_t)row*KH + 16 + m] = (half_t)s1;
    }
}

// ---------------------------------------------------------------------------
// G builders (small VALU GEMMs; f16 output = B-matrix of the message GEMMs).
// Rows [0,3072): G[j*32+k][o] = sum_d f0[j,d]*W2[k,d*CO+o]
// Rows [3072,3168): bias rows  = sum_d f0[j,d]*b2[d*CO+o]
// ---------------------------------------------------------------------------
__global__ void gbuild_f0_kernel(const float* __restrict__ f0,
                                 const float* __restrict__ w2,
                                 const float* __restrict__ b2,
                                 int in0, int CO, half_t* __restrict__ Gs)
{
    int b = blockIdx.y;
    int t = blockIdx.x * blockDim.x + threadIdx.x;
    if (t >= KB00 * CO) return;
    int row = t / CO, o = t % CO;
    const float* f0b = f0 + (size_t)b*NATOM*H0C;
    float acc = 0.f;
    if (row < KJK) {
        int j = row >> 5, k = row & 31;
        const float* wk = w2 + (size_t)k * in0 * CO;
        for (int d = 0; d < in0; ++d) acc += f0b[j*in0 + d] * wk[d*CO + o];
    } else {
        int j = row - KJK;
        for (int d = 0; d < in0; ++d) acc += f0b[j*in0 + d] * b2[d*CO + o];
    }
    Gs[(size_t)b*KB00*CO + t] = (half_t)acc;
}

// G10[(j*32+k)*3+x][o] = sum_c f1[j,c,x]*W2[k,c*64+o]; +288 bias rows (j,x)
__global__ void gbuild10_kernel(const float* __restrict__ f1,
                                const float* __restrict__ w2,
                                const float* __restrict__ b2,
                                int in1, half_t* __restrict__ Gs)
{
    int b = blockIdx.y;
    int t = blockIdx.x * blockDim.x + threadIdx.x;
    if (t >= KB10 * H0C) return;
    int row = t / H0C, o = t % H0C;
    const float* f1b = f1 + (size_t)b*NATOM*H1C*3;
    float acc = 0.f;
    if (row < NATOM*KH*3) {
        int j = row / (KH*3), u = row % (KH*3), k = u/3, x = u%3;
        const float* wk = w2 + (size_t)k * in1 * H0C;
        for (int cc = 0; cc < in1; ++cc)
            acc += f1b[(j*in1 + cc)*3 + x] * wk[cc*H0C + o];
    } else {
        int u = row - NATOM*KH*3, j = u/3, x = u%3;
        for (int cc = 0; cc < in1; ++cc)
            acc += f1b[(j*in1 + cc)*3 + x] * b2[cc*H0C + o];
    }
    Gs[(size_t)b*KB10*H0C + t] = (half_t)acc;
}

// G11[j*32+k][g*3+x] = sum_c f1[j,c,x]*W2[k,c*16+g]; +96 bias rows
__global__ void gbuild11_kernel(const float* __restrict__ f1,
                                const float* __restrict__ w2,
                                const float* __restrict__ b2,
                                int in1, half_t* __restrict__ Gs)
{
    int b = blockIdx.y;
    int t = blockIdx.x * blockDim.x + threadIdx.x;
    if (t >= KB00 * 48) return;
    int row = t / 48, col = t % 48, g = col/3, x = col%3;
    const float* f1b = f1 + (size_t)b*NATOM*H1C*3;
    float acc = 0.f;
    if (row < KJK) {
        int j = row >> 5, k = row & 31;
        const float* wk = w2 + (size_t)k * in1 * H1C;
        for (int cc = 0; cc < in1; ++cc)
            acc += f1b[(j*in1 + cc)*3 + x] * wk[cc*H1C + g];
    } else {
        int j = row - KJK;
        for (int cc = 0; cc < in1; ++cc)
            acc += f1b[(j*in1 + cc)*3 + x] * b2[cc*H1C + g];
    }
    Gs[(size_t)b*KB00*48 + t] = (half_t)acc;
}

// ---------------------------------------------------------------------------
// Message GEMM:  Out[i,col] (+)= sum_t A[i,t] * Bmat[t,col]
//   rmode 0: A = mask*h[i,j,k]           (t=(j,k); bias rows t>=3072: A=mask)
//   rmode 1: A = mask*h[i,j,k]*r[i,j,x]  (t=(j,k,x); bias rows: mask*r)
//   rmode 2: A = mask*h[i,j,k]*r[i,j,c]  (fixed component cfix)
// One wave per 16-row M tile; all NT 16-col tiles live in registers, so the
// masked A fragment is built once per K step and feeds NT back-to-back WMMAs.
// ---------------------------------------------------------------------------
template <int NT>
__global__ __launch_bounds__(32) void msg_gemm_kernel(
    const half_t* __restrict__ H, const float* __restrict__ rhat,
    const half_t* __restrict__ Bmat, float* __restrict__ Out,
    int Krows, int rmode, int cfix,
    int out_stride, int out_off, int accumulate)
{
    constexpr int NCOLS = NT * 16;
    __shared__ half_t As[16][32];
    __shared__ half_t Bs[32][NCOLS];
    int lane = threadIdx.x;
    int bx = blockIdx.x;          // M tile (0..5)
    int b  = blockIdx.z;          // batch
    const half_t* Hb = H + (size_t)b*NPAIR*KH;
    const float*  Rb = rhat + (size_t)b*NPAIR*3;
    const half_t* Bt = Bmat + (size_t)b*Krows*NCOLS;
    int m = lane & 15, hi = lane >> 4;
    v8f acc[NT];
#pragma unroll
    for (int nt = 0; nt < NT; ++nt) acc[nt] = (v8f){};
    int ktiles = Krows >> 5;
    for (int kt = 0; kt < ktiles; ++kt) {
#pragma unroll
        for (int e = 0; e < 16; ++e) {
            int q = lane*16 + e;
            int r = q >> 5, kk = q & 31;
            int i = bx*16 + r;
            int t = kt*32 + kk;
            float a;
            if (rmode == 1) {
                if (t < NATOM*KH*3) {
                    int j = t/(KH*3), u = t%(KH*3), k = u/3, x = u%3;
                    a = (i == j) ? 0.f
                        : (float)Hb[(size_t)(i*NATOM + j)*KH + k] *
                          Rb[(size_t)(i*NATOM + j)*3 + x];
                } else {
                    int u = t - NATOM*KH*3, j = u/3, x = u%3;
                    a = (i == j) ? 0.f : Rb[(size_t)(i*NATOM + j)*3 + x];
                }
            } else {
                int j, k = 0; bool biasrow;
                if (t < KJK) { j = t >> 5; k = t & 31; biasrow = false; }
                else         { j = t - KJK; biasrow = true; }
                if (i == j) a = 0.f;
                else {
                    float hv = biasrow ? 1.f
                              : (float)Hb[(size_t)(i*NATOM + j)*KH + k];
                    float rm = (rmode == 2)
                              ? Rb[(size_t)(i*NATOM + j)*3 + cfix] : 1.f;
                    a = hv * rm;
                }
            }
            As[r][kk] = (half_t)a;
        }
#pragma unroll
        for (int e = 0; e < 16*NT; ++e) {
            int q = lane*(16*NT) + e;
            int kk = q / NCOLS, n = q % NCOLS;
            Bs[kk][n] = Bt[(size_t)(kt*32 + kk)*NCOLS + n];
        }
        __syncthreads();
        v16h a;
#pragma unroll
        for (int e = 0; e < 16; ++e) {
            int k = e + 16*hi;
            a[e] = As[m][k];
        }
#pragma unroll
        for (int nt = 0; nt < NT; ++nt) {
            v16h bf;
#pragma unroll
            for (int e = 0; e < 16; ++e) bf[e] = Bs[e + 16*hi][nt*16 + m];
            acc[nt] = __builtin_amdgcn_wmma_f32_16x16x32_f16(
                false, a, false, bf, (short)0, acc[nt], false, false);
        }
        __syncthreads();
    }
#pragma unroll
    for (int nt = 0; nt < NT; ++nt) {
#pragma unroll
        for (int v = 0; v < 8; ++v) {
            int row = bx*16 + v + 8*hi;
            int col = nt*16 + m;
            size_t idx = ((size_t)(b*NATOM + row)*NCOLS + col)*out_stride
                       + out_off;
            if (accumulate) Out[idx] += acc[nt][v]; else Out[idx] = acc[nt][v];
        }
    }
}

// ---------------------------------------------------------------------------
// f0 layernorm (64 wide), one wave per row, wave32 shuffle reduction.
// ---------------------------------------------------------------------------
__global__ __launch_bounds__(32) void ln_f0_kernel(
    const float* __restrict__ msg0, const float* __restrict__ gam,
    const float* __restrict__ bet, float* __restrict__ f0out)
{
    int rowid = blockIdx.x;               // b*96 + i
    int lane  = threadIdx.x;
    const float* x = msg0 + (size_t)rowid*H0C;
    float x0 = x[lane], x1 = x[lane + 32];
    float s = x0 + x1;
#pragma unroll
    for (int msk = 16; msk >= 1; msk >>= 1) s += __shfl_xor(s, msk, 32);
    float mean = s * (1.0f/64.0f);
    float d0 = x0 - mean, d1 = x1 - mean;
    float vv = d0*d0 + d1*d1;
#pragma unroll
    for (int msk = 16; msk >= 1; msk >>= 1) vv += __shfl_xor(vv, msk, 32);
    float inv = rsqrtf(vv*(1.0f/64.0f) + 1e-5f);
    f0out[(size_t)rowid*H0C + lane]      = d0*inv*gam[lane]      + bet[lane];
    f0out[(size_t)rowid*H0C + lane + 32] = d1*inv*gam[lane + 32] + bet[lane + 32];
}

// f1 gating: norms -> clip -> LN over H1 -> rescale direction vectors.
__global__ __launch_bounds__(32) void f1_norm_kernel(
    const float* __restrict__ msg1, const float* __restrict__ gam,
    const float* __restrict__ bet, float* __restrict__ f1out)
{
    int rowid = blockIdx.x;
    int lane  = threadIdx.x;
    const float* raw = msg1 + (size_t)rowid*H1C*3;
    float r0 = 0.f, r1 = 0.f, r2 = 0.f, nc = 0.f;
    if (lane < H1C) {
        r0 = raw[lane*3+0]; r1 = raw[lane*3+1]; r2 = raw[lane*3+2];
        nc = fmaxf(sqrtf(r0*r0 + r1*r1 + r2*r2), 1e-8f);
    }
    float s = (lane < H1C) ? nc : 0.f;
#pragma unroll
    for (int msk = 16; msk >= 1; msk >>= 1) s += __shfl_xor(s, msk, 32);
    float mean = s * (1.0f/16.0f);
    float dv = (lane < H1C) ? (nc - mean) : 0.f;
    float vv = dv*dv;
#pragma unroll
    for (int msk = 16; msk >= 1; msk >>= 1) vv += __shfl_xor(vv, msk, 32);
    float inv = rsqrtf(vv*(1.0f/16.0f) + 1e-5f);
    if (lane < H1C) {
        float ln = (nc - mean)*inv*gam[lane] + bet[lane];
        float sc = ln * fast_rcp(nc);
        float* o = f1out + (size_t)rowid*H1C*3 + lane*3;
        o[0] = r0*sc; o[1] = r1*sc; o[2] = r2*sc;
    }
}

// feat[i] = [f0 | ||f1||]; max over atoms.
__global__ void feat_kernel(const float* __restrict__ f0,
                            const float* __restrict__ f1,
                            float* __restrict__ featmax)
{
    int b = blockIdx.x, o = threadIdx.x;
    if (o >= H0C + H1C) return;
    float m = -3.0e38f;
    for (int i = 0; i < NATOM; ++i) {
        float v;
        if (o < H0C) v = f0[(size_t)(b*NATOM + i)*H0C + o];
        else {
            const float* r = f1 + ((size_t)(b*NATOM + i)*H1C + (o - H0C))*3;
            v = sqrtf(r[0]*r[0] + r[1]*r[1] + r[2]*r[2]);
        }
        m = fmaxf(m, v);
    }
    featmax[b*(H0C + H1C) + o] = m;
}

// classifier head 80 -> 128 -> 64 -> 10 (tiny; LDS + VALU)
__global__ __launch_bounds__(128) void rho_kernel(
    const float* __restrict__ featmax,
    const float* __restrict__ w0, const float* __restrict__ bb0,
    const float* __restrict__ w1, const float* __restrict__ bb1,
    const float* __restrict__ w2, const float* __restrict__ bb2,
    float* __restrict__ out)
{
    __shared__ float x0[80], x1[128], x2[64];
    int b = blockIdx.x, t = threadIdx.x;
    if (t < 80) x0[t] = featmax[b*80 + t];
    __syncthreads();
    if (t < 128) {
        float a = bb0[t];
        for (int k = 0; k < 80; ++k) a += x0[k]*w0[k*128 + t];
        x1[t] = fmaxf(a, 0.f);
    }
    __syncthreads();
    if (t < 64) {
        float a = bb1[t];
        for (int k = 0; k < 128; ++k) a += x1[k]*w1[k*64 + t];
        x2[t] = fmaxf(a, 0.f);
    }
    __syncthreads();
    if (t < 10) {
        float a = bb2[t];
        for (int k = 0; k < 64; ++k) a += x2[k]*w2[k*10 + t];
        out[b*10 + t] = a;
    }
}

// ---------------------------------------------------------------------------
extern "C" void kernel_launch(void* const* d_in, const int* in_sizes, int n_in,
                              void* d_out, int out_size, void* d_ws, size_t ws_size,
                              hipStream_t stream)
{
    (void)out_size; (void)ws_size;
    auto F = [&](int i) -> const float* { return (const float*)d_in[i]; };

    // Detect pytree flattening order.
    bool pos_first = (n_in > 0 && in_sizes[0] == BATCH*NATOM*3);
    const float* pos;
    // filter index: 0=W00, 1=W10, 2=W01, 3=W11
    const float *w1f[3][4], *b1f[3][4], *w2f[3][4], *b2f[3][4];
    const float *g0p[3], *b0p[3], *g1p[3], *b1p[3];
    const float *rw[3], *rb[3];

    if (!pos_first) {
        // jax-sorted: per-layer W00,W01,W10,W11 each (l1.b,l1.w,l2.b,l2.w),
        // then b0,b1,g0,g1; rho lins (b,w); pos last.
        pos = F(66);
        const int off[4] = {0, 8, 4, 12};     // {W00,W10,W01,W11} positions
        for (int li = 0; li < 3; ++li) {
            int base = li*20;
            for (int f = 0; f < 4; ++f) {
                int o = base + off[f];
                b1f[li][f] = F(o+0); w1f[li][f] = F(o+1);
                b2f[li][f] = F(o+2); w2f[li][f] = F(o+3);
            }
            b0p[li] = F(base+16); b1p[li] = F(base+17);
            g0p[li] = F(base+18); g1p[li] = F(base+19);
        }
        rb[0] = F(60); rw[0] = F(61); rb[1] = F(62);
        rw[1] = F(63); rb[2] = F(64); rw[2] = F(65);
    } else {
        // insertion order: pos, per-layer W00,W10,W01,W11 each (w,b,w,b),
        // g0,b0,g1,b1; rho lins (w,b).
        pos = F(0);
        const int off[4] = {0, 4, 8, 12};
        for (int li = 0; li < 3; ++li) {
            int base = 1 + li*20;
            for (int f = 0; f < 4; ++f) {
                int o = base + off[f];
                w1f[li][f] = F(o+0); b1f[li][f] = F(o+1);
                w2f[li][f] = F(o+2); b2f[li][f] = F(o+3);
            }
            g0p[li] = F(base+16); b0p[li] = F(base+17);
            g1p[li] = F(base+18); b1p[li] = F(base+19);
        }
        rw[0] = F(61); rb[0] = F(62); rw[1] = F(63);
        rb[1] = F(64); rw[2] = F(65); rb[2] = F(66);
    }

    // Workspace carve (~10 MB total).
    char* wsp = (char*)d_ws;
    size_t offb = 0;
    auto alloc = [&](size_t bytes) -> void* {
        void* p = wsp + offb;
        offb = (offb + bytes + 255) & ~(size_t)255;
        return p;
    };
    float*  rhat = (float*) alloc((size_t)BATCH*NPAIR*3*4);
    half_t* rbf  = (half_t*)alloc((size_t)BATCH*NPAIR*NRBF*2);
    half_t* h[4];
    for (int f = 0; f < 4; ++f) h[f] = (half_t*)alloc((size_t)BATCH*NPAIR*KH*2);
    half_t* G00s = (half_t*)alloc((size_t)BATCH*KB00*64*2);
    half_t* G10s = (half_t*)alloc((size_t)BATCH*KB10*64*2);
    half_t* G01s = (half_t*)alloc((size_t)BATCH*KB00*16*2);
    half_t* G11s = (half_t*)alloc((size_t)BATCH*KB00*48*2);
    float*  f0A  = (float*) alloc((size_t)BATCH*NATOM*H0C*4);
    float*  f0B  = (float*) alloc((size_t)BATCH*NATOM*H0C*4);
    float*  f1A  = (float*) alloc((size_t)BATCH*NATOM*H1C*3*4);
    float*  f1B  = (float*) alloc((size_t)BATCH*NATOM*H1C*3*4);
    float*  msg0 = (float*) alloc((size_t)BATCH*NATOM*H0C*4);
    float*  msg1 = (float*) alloc((size_t)BATCH*NATOM*H1C*3*4);
    float*  fmax = (float*) alloc((size_t)BATCH*80*4);

    geom_kernel<<<dim3((BATCH*NPAIR + 255)/256), 256, 0, stream>>>(
        pos, rhat, rbf, f0A, f1A);

    float *f0cur = f0A, *f0nxt = f0B, *f1cur = f1A, *f1nxt = f1B;
    int in0 = 1, in1 = 1;
    for (int li = 0; li < 3; ++li) {
        for (int f = 0; f < 4; ++f)
            filter_h_kernel<<<dim3(BATCH*NPAIR/16), 32, 0, stream>>>(
                rbf, w1f[li][f], b1f[li][f], h[f]);

        gbuild_f0_kernel<<<dim3((KB00*64 + 255)/256, BATCH), 256, 0, stream>>>(
            f0cur, w2f[li][0], b2f[li][0], in0, 64, G00s);
        gbuild_f0_kernel<<<dim3((KB00*16 + 255)/256, BATCH), 256, 0, stream>>>(
            f0cur, w2f[li][2], b2f[li][2], in0, 16, G01s);
        gbuild10_kernel<<<dim3((KB10*64 + 255)/256, BATCH), 256, 0, stream>>>(
            f1cur, w2f[li][1], b2f[li][1], in1, G10s);
        gbuild11_kernel<<<dim3((KB00*48 + 255)/256, BATCH), 256, 0, stream>>>(
            f1cur, w2f[li][3], b2f[li][3], in1, G11s);

        // msg00: write, msg10: accumulate  -> msg0 (N x 64)
        msg_gemm_kernel<4><<<dim3(6, 1, BATCH), 32, 0, stream>>>(
            h[0], rhat, G00s, msg0, KB00, 0, 0, 1, 0, 0);
        msg_gemm_kernel<4><<<dim3(6, 1, BATCH), 32, 0, stream>>>(
            h[1], rhat, G10s, msg0, KB10, 1, 0, 1, 0, 1);
        // msg01 (3 vector components, strided writes), msg11 accumulate -> msg1
        for (int comp = 0; comp < 3; ++comp)
            msg_gemm_kernel<1><<<dim3(6, 1, BATCH), 32, 0, stream>>>(
                h[2], rhat, G01s, msg1, KB00, 2, comp, 3, comp, 0);
        msg_gemm_kernel<3><<<dim3(6, 1, BATCH), 32, 0, stream>>>(
            h[3], rhat, G11s, msg1, KB00, 0, 0, 1, 0, 1);

        ln_f0_kernel<<<BATCH*NATOM, 32, 0, stream>>>(msg0, g0p[li], b0p[li], f0nxt);
        f1_norm_kernel<<<BATCH*NATOM, 32, 0, stream>>>(msg1, g1p[li], b1p[li], f1nxt);

        float* t;
        t = f0cur; f0cur = f0nxt; f0nxt = t;
        t = f1cur; f1cur = f1nxt; f1nxt = t;
        in0 = H0C; in1 = H1C;
    }

    feat_kernel<<<BATCH, 128, 0, stream>>>(f0cur, f1cur, fmax);
    rho_kernel<<<BATCH, 128, 0, stream>>>(fmax, rw[0], rb[0], rw[1], rb[1],
                                          rw[2], rb[2], (float*)d_out);
}
